// RNAPath_52888227283780
// MI455X (gfx1250) — matrix operand order repaired and checked by reading, not compiled, not run on hardware
//
#include <hip/hip_runtime.h>

// ---------------- problem constants ----------------
#define K_PATCHES 16384
#define D_FEAT    384
#define N_GENES   20000
#define N_PAD     20224                     // 79 * 256 (gene dim padded for tiling)
#define NT_BLOCKS 79                        // N_PAD / 256
#define KT        128                       // patch rows per K-chunk (WG K tile)
#define NTILE     256                       // genes per workgroup
#define KBANDS    16                        // grid.y
#define CHUNKS    (K_PATCHES / (KT * KBANDS))   // 8 K-chunks per band
#define DSTEP     32                        // WMMA K depth per step (bf16)
#define DSTEPS    (D_FEAT / DSTEP)          // 12

typedef __attribute__((ext_vector_type(16))) __bf16 v16bf;
typedef __attribute__((ext_vector_type(8)))  float  v8f;
typedef unsigned int   u32;
typedef unsigned short u16;
typedef __attribute__((ext_vector_type(4))) u32 u32x4;
typedef __attribute__((ext_vector_type(8))) u32 u32x8;

// TDM descriptor group types (probe: 5-arg on clang-22/ROCm7.2, 6-arg on clang-23)
typedef __attribute__((ext_vector_type(4))) u32 tdm_g0_t;
typedef __attribute__((ext_vector_type(8))) int tdm_g1_t;
typedef __attribute__((ext_vector_type(4))) int tdm_g2_t;
typedef __attribute__((ext_vector_type(8))) int tdm_g3x_t;

__device__ __forceinline__ u16 f32_to_bf16(float f) {
    u32 u = __float_as_uint(f);
    u32 r = u + 0x7FFFu + ((u >> 16) & 1u);   // round-to-nearest-even
    return (u16)(r >> 16);
}

__device__ __forceinline__ v16bf make_frag(u32x4 lo, u32x4 hi) {
    u32x8 t;
    t[0] = lo[0]; t[1] = lo[1]; t[2] = lo[2]; t[3] = lo[3];
    t[4] = hi[0]; t[5] = hi[1]; t[6] = hi[2]; t[7] = hi[3];
    return __builtin_bit_cast(v16bf, t);
}

// Issue one TDM 2-D tile load: (tile1 rows) x (tile0 elems) of 2-byte data,
// row stride = stride0 elems, global tile start gptr -> LDS byte offset lds_off.
// D# layout per CDNA5 ISA.
__device__ __forceinline__ void tdm_load_2d(u32 lds_off, const void* gptr,
                                            u32 tdim0, u32 tdim1,
                                            u32 tile0, u32 tile1, u32 stride0) {
    unsigned long long ga = (unsigned long long)gptr;
    tdm_g0_t g0;
    g0[0] = 1u;                               // count=1, user mode, no gather
    g0[1] = lds_off;                          // lds_addr (bytes)
    g0[2] = (u32)ga;                          // global_addr[31:0]
    g0[3] = (u32)(ga >> 32) | (2u << 30);     // global_addr[56:32] | type=2
    tdm_g1_t g1;
    g1[0] = 0x00010000;                       // workgroup_mask=0, data_size=1 (2B)
    g1[1] = (int)((tdim0 & 0xFFFFu) << 16);                       // tensor_dim0 lo
    g1[2] = (int)((tdim0 >> 16) | ((tdim1 & 0xFFFFu) << 16));     // dim0 hi | dim1 lo
    g1[3] = (int)((tdim1 >> 16) | (tile0 << 16));                 // dim1 hi | tile_dim0
    g1[4] = (int)tile1;                       // tile_dim1 | tile_dim2=0
    g1[5] = (int)stride0;                     // tensor_dim0_stride[31:0]
    g1[6] = 0;                                // stride0 hi | stride1 lo (unused, 2-D)
    g1[7] = 0;
    tdm_g2_t z4 = {};
#if __clang_major__ >= 23
    tdm_g3x_t z8 = {};
    __builtin_amdgcn_tensor_load_to_lds(g0, g1, z4, z4, z8, 0);
#else
    __builtin_amdgcn_tensor_load_to_lds(g0, g1, z4, z4, 0);
#endif
}

// Pass 1: fp32 -> bf16 for h and W (W zero-padded to N_PAD rows), zero d_out.
__global__ void rnapath_cvt_zero(const float* __restrict__ h,
                                 const float* __restrict__ W,
                                 float* __restrict__ out,
                                 u16* __restrict__ hb,
                                 u16* __restrict__ wb) {
    int i = blockIdx.x * blockDim.x + threadIdx.x;
    if (i < K_PATCHES * D_FEAT) hb[i] = f32_to_bf16(h[i]);
    if (i < N_PAD * D_FEAT)
        wb[i] = (i < N_GENES * D_FEAT) ? f32_to_bf16(W[i]) : (u16)0;
    if (i < N_GENES) out[i] = 0.0f;
}

// Pass 2: bf16 WMMA GEMM with fused bias + ReLU + mean-over-K reduction.
// WG = 256 threads = 8 waves in a 2(K) x 4(N) grid; wave tile = 64 x 64.
// Tiles are DMA'd into double-buffered LDS by the Tensor Data Mover (wave 0),
// synchronized with s_wait_tensorcnt + workgroup barrier; the next step's DMA
// is in flight while the current step's 16 WMMAs execute.
__global__ __launch_bounds__(256)
void rnapath_gemm_relu_mean(const u16* __restrict__ hb,   // [K_PATCHES][D_FEAT] bf16
                            const u16* __restrict__ wb,   // [N_PAD][D_FEAT] bf16
                            const float* __restrict__ bias,
                            float* __restrict__ out) {
    __shared__ u16 As[2][KT * DSTEP];      // 2 x 8 KB   [patch][d]
    __shared__ u16 Bs[2][NTILE * DSTEP];   // 2 x 16 KB  [gene][d]

    const int tid  = threadIdx.x;
    const int lane = tid & 31;
    const int wave = tid >> 5;          // 0..7
    const int wk   = wave >> 2;         // 0..1 : 64-row K half of the 128-row tile
    const int wn   = wave & 3;          // 0..3 : 64-col N quarter of the 256-col tile
    const int hl   = lane >> 4;         // half-lane group 0/1
    const int m    = lane & 15;

    const int n0   = blockIdx.x * NTILE;
    const int band = blockIdx.y;

    // Flat LDS byte offsets (generic addr [31:0] is the LDS offset, ISA §10.2).
    const u32 asOff0 = (u32)(size_t)(void*)&As[0][0];
    const u32 asOff1 = (u32)(size_t)(void*)&As[1][0];
    const u32 bsOff0 = (u32)(size_t)(void*)&Bs[0][0];
    const u32 bsOff1 = (u32)(size_t)(void*)&Bs[1][0];

    // Per-lane gene columns for this wave's four 16-wide N subtiles.
    int   g[4];
    float bn[4];
    float part[4];
#pragma unroll
    for (int nt = 0; nt < 4; ++nt) {
        g[nt]    = n0 + wn * 64 + nt * 16 + m;
        bn[nt]   = (g[nt] < N_GENES) ? bias[g[nt]] : 0.0f;
        part[nt] = 0.0f;
    }

    // Prologue: DMA step 0 tiles into buffer 0.
    if (wave == 0) {
        const int k0 = band * CHUNKS * KT;
        tdm_load_2d(asOff0, hb + (size_t)k0 * D_FEAT, D_FEAT, K_PATCHES,
                    DSTEP, KT, D_FEAT);
        tdm_load_2d(bsOff0, wb + (size_t)n0 * D_FEAT, D_FEAT, N_PAD,
                    DSTEP, NTILE, D_FEAT);
    }

    int s = 0;                               // linear step index (0..95)
    for (int c = 0; c < CHUNKS; ++c) {
        v8f acc[4][4];
#pragma unroll
        for (int mt = 0; mt < 4; ++mt)
#pragma unroll
            for (int nt = 0; nt < 4; ++nt) acc[mt][nt] = {};

        for (int ds = 0; ds < DSTEPS; ++ds, ++s) {
            const int p = s & 1;
            // Step-s DMA complete (wave 0's TENSORcnt; no-op for other waves).
            __builtin_amdgcn_s_wait_tensorcnt(0);
            // Publish buffer p to all waves; also guarantees every wave has
            // finished reading buffer p^1 (its ds_loads completed before the
            // WMMAs of step s-1), so p^1 is free for the next DMA.
            __syncthreads();

            // Kick off DMA for step s+1 into the other buffer (overlaps WMMAs).
            if (wave == 0 && s + 1 < CHUNKS * DSTEPS) {
                int cn = c, dsn = ds + 1;
                if (dsn == DSTEPS) { dsn = 0; cn = c + 1; }
                const int kn  = (band * CHUNKS + cn) * KT;
                const int dbn = dsn * DSTEP;
                tdm_load_2d(p ? asOff0 : asOff1,
                            hb + (size_t)kn * D_FEAT + dbn,
                            D_FEAT, K_PATCHES, DSTEP, KT, D_FEAT);
                tdm_load_2d(p ? bsOff0 : bsOff1,
                            wb + (size_t)n0 * D_FEAT + dbn,
                            D_FEAT, N_PAD, DSTEP, NTILE, D_FEAT);
            }

            // A fragments: 16-bit A layout -> K(e) = (e&7) + 8*hl + 16*(e>>3)
            // == dwords {0..3}+4*hl and {8..11}+4*hl of the 16-dword row.
            v16bf afr[4];
#pragma unroll
            for (int mt = 0; mt < 4; ++mt) {
                const u32x4* Ar = (const u32x4*)&As[p][(wk * 64 + mt * 16 + m) * DSTEP];
                afr[mt] = make_frag(Ar[hl], Ar[2 + hl]);
            }
            // B fragments: same K order as A -> K(e) = (e&7) + 8*hl + 16*(e>>3)
            // == dwords {0..3}+4*hl and {8..11}+4*hl of the 16-dword row.
            v16bf bfr[4];
#pragma unroll
            for (int nt = 0; nt < 4; ++nt) {
                const u32x4* Br = (const u32x4*)&Bs[p][(wn * 64 + nt * 16 + m) * DSTEP];
                bfr[nt] = make_frag(Br[hl], Br[2 + hl]);
            }

#pragma unroll
            for (int mt = 0; mt < 4; ++mt)
#pragma unroll
                for (int nt = 0; nt < 4; ++nt)
                    acc[mt][nt] = __builtin_amdgcn_wmma_f32_16x16x32_bf16(
                        false, afr[mt], false, bfr[nt],
                        (short)0, acc[mt][nt], false, false);
        }

        // Fused epilogue: bias + ReLU + sum over this chunk's K rows.
        // C layout: VGPR r -> M=r (lanes 0-15) / M=r+8 (lanes 16-31), lane -> N.
#pragma unroll
        for (int nt = 0; nt < 4; ++nt) {
#pragma unroll
            for (int mt = 0; mt < 4; ++mt) {
#pragma unroll
                for (int r = 0; r < 8; ++r) {
                    float v = acc[mt][nt][r] + bn[nt];
                    part[nt] += (v > 0.0f) ? v : 0.0f;
                }
            }
        }
    }

    // Combine the two half-lane groups (rows 0-7 vs 8-15 of each C tile),
    // then one scaled atomic per gene column.
    const float scale = 1.0f / (float)K_PATCHES;   // mean over patches
#pragma unroll
    for (int nt = 0; nt < 4; ++nt) {
        part[nt] += __shfl_xor(part[nt], 16, 32);
        if (lane < 16 && g[nt] < N_GENES)
            atomicAdd(&out[g[nt]], part[nt] * scale);
    }
}

extern "C" void kernel_launch(void* const* d_in, const int* in_sizes, int n_in,
                              void* d_out, int out_size, void* d_ws, size_t ws_size,
                              hipStream_t stream) {
    const float* h = (const float*)d_in[0];   // (16384, 384) f32
    const float* W = (const float*)d_in[1];   // (20000, 384) f32
    const float* b = (const float*)d_in[2];   // (20000,)     f32
    float* out = (float*)d_out;               // (20000,)     f32

    const size_t hbBytes = (size_t)K_PATCHES * D_FEAT * sizeof(u16);   // 12.6 MB
    const size_t wbBytes = (size_t)N_PAD * D_FEAT * sizeof(u16);       // 15.5 MB
    if (ws_size < hbBytes + wbBytes) return;
    u16* hb = (u16*)d_ws;
    u16* wb = (u16*)((char*)d_ws + hbBytes);

    const int total = N_PAD * D_FEAT;
    rnapath_cvt_zero<<<(total + 255) / 256, 256, 0, stream>>>(h, W, out, hb, wb);

    dim3 grid(NT_BLOCKS, KBANDS);   // 79 x 16 = 1264 workgroups
    rnapath_gemm_relu_mean<<<grid, 256, 0, stream>>>(hb, wb, b, out);
}
